// draw_44727789421007
// MI455X (gfx1250) — matrix operand order repaired and checked by reading, not compiled, not run on hardware
//
#include <hip/hip_runtime.h>
#include <hip/hip_bf16.h>
#include <cstdint>
#include <cstddef>

// ---------------- constants ----------------
#define BSZ     2048
#define T_STEPS 16
#define NPATCH  16          // N
#define AW      32          // A == B == IMG
#define CHW     1024        // 32*32
#define IMGE    3072        // 3*32*32
#define HDIM    100         // ENC_H == DEC_H == NZ
#define IN_ENC  1636
#define K_ENC   1664        // IN_ENC padded to mult of 32
#define K_H     128         // 100 padded to mult of 32
#define N_G3    320         // 300 padded to mult of 32
#define N_G4    416         // 400 padded to mult of 32
#define N_FC    128         // 100 padded to mult of 32
#define N_WR    768         // already mult of 32

typedef __attribute__((ext_vector_type(16))) __bf16 v16bf;
typedef __attribute__((ext_vector_type(8)))  __bf16 v8bf;
typedef __attribute__((ext_vector_type(8)))  float  v8f;

__device__ __forceinline__ float sigf(float x) { return 1.0f / (1.0f + expf(-x)); }

// ---------------- utility kernels ----------------
__global__ void zero_u32_kernel(unsigned* p, size_t n) {
    size_t i = (size_t)blockIdx.x * blockDim.x + threadIdx.x;
    size_t stride = (size_t)gridDim.x * blockDim.x;
    for (; i < n; i += stride) p[i] = 0u;
}

__global__ void convert_pad_kernel(const float* __restrict__ src, __bf16* __restrict__ dst,
                                   int N, int K, int Npad, int Kpad) {
    size_t idx = (size_t)blockIdx.x * blockDim.x + threadIdx.x;
    size_t tot = (size_t)Npad * Kpad;
    size_t stride = (size_t)gridDim.x * blockDim.x;
    for (; idx < tot; idx += stride) {
        int n = (int)(idx / Kpad), k = (int)(idx % Kpad);
        float v = (n < N && k < K) ? src[(size_t)n * K + k] : 0.0f;
        dst[idx] = (__bf16)v;
    }
}

// ---------------- WMMA GEMM with 2x2 register blocking ----------------
// C[M x Npad] = A_bf16[M x Kpad] * W_bf16[Npad x Kpad]^T + bias
// Each wave computes a 32x32 output block: 4 accumulators, A fragments reused
// across two N-tiles and B fragments across two M-tiles (4 WMMA per 8 b128 loads).
__device__ __forceinline__ void gemm_kstep(const __bf16* rowA0, const __bf16* rowA1,
                                           const __bf16* rowW0, const __bf16* rowW1,
                                           int k0, int kh,
                                           v8f& acc00, v8f& acc01, v8f& acc10, v8f& acc11) {
    union { v16bf v; v8bf h[2]; } a0, a1;
    a0.h[0] = *(const v8bf*)(rowA0 + k0 + kh * 8);       // K = k0 + kh*8 + [0..7]
    a0.h[1] = *(const v8bf*)(rowA0 + k0 + 16 + kh * 8);  // K = k0+16+kh*8 + [0..7]
    a1.h[0] = *(const v8bf*)(rowA1 + k0 + kh * 8);
    a1.h[1] = *(const v8bf*)(rowA1 + k0 + 16 + kh * 8);
    union { v16bf v; v8bf h[2]; } b0, b1;
    b0.h[0] = *(const v8bf*)(rowW0 + k0 + kh * 8);       // K = k0 + kh*8 + [0..7]
    b0.h[1] = *(const v8bf*)(rowW0 + k0 + 16 + kh * 8);  // K = k0+16+kh*8 + [0..7]
    b1.h[0] = *(const v8bf*)(rowW1 + k0 + kh * 8);
    b1.h[1] = *(const v8bf*)(rowW1 + k0 + 16 + kh * 8);
    acc00 = __builtin_amdgcn_wmma_f32_16x16x32_bf16(false, a0.v, false, b0.v, (short)0, acc00, false, false);
    acc01 = __builtin_amdgcn_wmma_f32_16x16x32_bf16(false, a0.v, false, b1.v, (short)0, acc01, false, false);
    acc10 = __builtin_amdgcn_wmma_f32_16x16x32_bf16(false, a1.v, false, b0.v, (short)0, acc10, false, false);
    acc11 = __builtin_amdgcn_wmma_f32_16x16x32_bf16(false, a1.v, false, b1.v, (short)0, acc11, false, false);
}

__global__ void wmma_gemm_kernel(const __bf16* __restrict__ A, int lda,
                                 const __bf16* __restrict__ W, int ldw,
                                 const float* __restrict__ bias, int Nreal,
                                 float* __restrict__ C, int ldc,
                                 int M, int Npad, int Kpad) {
    int wave = threadIdx.x >> 5;
    int lane = threadIdx.x & 31;
    int tile = blockIdx.x * (blockDim.x >> 5) + wave;
    int ntn  = Npad >> 5;                       // number of 32-wide N blocks
    int mt = tile / ntn, nt = tile % ntn;
    if (mt * 32 >= M) return;                   // uniform per wave
    int m0 = mt << 5, n0 = nt << 5;
    int l15 = lane & 15, kh = lane >> 4;        // kh selects K half per ISA layout

    const __bf16* rowA0 = A + (size_t)(m0 + l15) * lda;        // A rows m0+l15
    const __bf16* rowA1 = A + (size_t)(m0 + 16 + l15) * lda;   // A rows m0+16+l15
    const __bf16* rowW0 = W + (size_t)(n0 + l15) * ldw;        // B cols n0+l15
    const __bf16* rowW1 = W + (size_t)(n0 + 16 + l15) * ldw;   // B cols n0+16+l15

    v8f acc00 = {}, acc01 = {}, acc10 = {}, acc11 = {};
    // main loop: unconditional prefetch of next K chunk (in-bounds by construction)
    int k0 = 0;
    for (; k0 < Kpad - 32; k0 += 32) {
        gemm_kstep(rowA0, rowA1, rowW0, rowW1, k0, kh, acc00, acc01, acc10, acc11);
        __builtin_prefetch(rowA0 + k0 + 32, 0, 3);   // global_prefetch_b8, near-cache
        __builtin_prefetch(rowA1 + k0 + 32, 0, 3);
        __builtin_prefetch(rowW0 + k0 + 32, 0, 3);
        __builtin_prefetch(rowW1 + k0 + 32, 0, 3);
    }
    // peeled last iteration (no prefetch)
    gemm_kstep(rowA0, rowA1, rowW0, rowW1, k0, kh, acc00, acc01, acc10, acc11);

    float bv0 = (bias != nullptr && (n0 + l15) < Nreal)      ? bias[n0 + l15]      : 0.0f;
    float bv1 = (bias != nullptr && (n0 + 16 + l15) < Nreal) ? bias[n0 + 16 + l15] : 0.0f;
#pragma unroll
    for (int r = 0; r < 8; ++r) {
        // C layout: v[r] -> M = r + 8*(lane>>4), N = lane&15 within each 16x16 tile
        size_t mr0 = (size_t)(m0 + (kh << 3) + r) * ldc;
        size_t mr1 = (size_t)(m0 + 16 + (kh << 3) + r) * ldc;
        C[mr0 + n0 + l15]      = acc00[r] + bv0;
        C[mr0 + n0 + 16 + l15] = acc01[r] + bv1;
        C[mr1 + n0 + l15]      = acc10[r] + bv0;
        C[mr1 + n0 + 16 + l15] = acc11[r] + bv1;
    }
}

// ---------------- attention filters ----------------
__global__ void attn_kernel(const float* __restrict__ h_dec, const float* __restrict__ W_att,
                            const float* __restrict__ b_att,
                            float* __restrict__ Fx, float* __restrict__ Fy,
                            float* __restrict__ gam) {
    int b = blockIdx.x * blockDim.x + threadIdx.x;
    if (b >= BSZ) return;
    const float* h = h_dec + (size_t)b * HDIM;
    float p[5];
#pragma unroll
    for (int k = 0; k < 5; ++k) {
        float acc = b_att[k];
        const float* w = W_att + k * HDIM;
        for (int j = 0; j < HDIM; ++j) acc += h[j] * w[j];
        p[k] = acc;
    }
    float gx = (AW + 1) * (p[0] + 1.0f) * 0.5f;
    float gy = (AW + 1) * (p[1] + 1.0f) * 0.5f;
    float var = expf(p[2]);
    float delta = ((float)(AW - 1) / (float)(NPATCH - 1)) * expf(p[3]);
    gam[b] = expf(p[4]);
    float inv2v = 1.0f / (2.0f * var);
    for (int n = 0; n < NPATCH; ++n) {
        float mui = gx + ((float)n - (NPATCH * 0.5f + 0.5f)) * delta;
        float muj = gy + ((float)n - (NPATCH * 0.5f + 0.5f)) * delta;
        float fx[AW];
        float s = 0.0f;
#pragma unroll
        for (int a = 0; a < AW; ++a) {
            float d = (float)a - mui;
            float v = expf(-d * d * inv2v);
            fx[a] = v; s += v;
        }
        float inv = 1.0f / s;
        float* fxp = Fx + ((size_t)b * NPATCH + n) * AW;
        float* fyp = Fy + ((size_t)b * NPATCH + n) * AW;
#pragma unroll
        for (int a = 0; a < AW; ++a) fxp[a] = fx[a] * inv;
#pragma unroll
        for (int a = 0; a < AW; ++a) {             // faithful bug: Fy NOT normalized
            float d = (float)a - muj;
            fyp[a] = expf(-d * d * inv2v);
        }
    }
}

// ---------------- attention read: enc_in = [gamma * Fy x6 Fx^T (1536) , h_dec (100), pad0] ----------------
__global__ void read_kernel(const float* __restrict__ x, const float* __restrict__ canvas,
                            const float* __restrict__ Fx, const float* __restrict__ Fy,
                            const float* __restrict__ gam, const float* __restrict__ h_dec,
                            __bf16* __restrict__ enc_in) {
    __shared__ float sFx[NPATCH * AW];
    __shared__ float sFy[NPATCH * AW];
    __shared__ float sImg[AW * AW];
    __shared__ float sTmp[NPATCH * AW];
    int b = blockIdx.x, tid = threadIdx.x;
    for (int i = tid; i < NPATCH * AW; i += 256) {
        sFx[i] = Fx[(size_t)b * NPATCH * AW + i];
        sFy[i] = Fy[(size_t)b * NPATCH * AW + i];
    }
    if (tid < HDIM)
        enc_in[(size_t)b * K_ENC + 1536 + tid] = (__bf16)h_dec[(size_t)b * HDIM + tid];
    float g = gam[b];
    for (int ch = 0; ch < 6; ++ch) {
        __syncthreads();
        size_t base = (size_t)b * IMGE + (size_t)(ch % 3) * CHW;
        for (int i = tid; i < CHW; i += 256) {
            float v = x[base + i];
            if (ch >= 3) v -= sigf(canvas[base + i]);
            sImg[i] = v;
        }
        __syncthreads();
        for (int i = tid; i < NPATCH * AW; i += 256) {   // tmp[n][xc] = sum_y Fy[n][y]*img[y][xc]
            int n = i >> 5, xc = i & 31;
            float acc = 0.0f;
#pragma unroll
            for (int y = 0; y < AW; ++y) acc += sFy[n * AW + y] * sImg[y * AW + xc];
            sTmp[i] = acc;
        }
        __syncthreads();
        {   // rd[n][m] = g * sum_x tmp[n][x]*Fx[m][x]
            int n = tid >> 4, m = tid & 15;
            float acc = 0.0f;
#pragma unroll
            for (int xx = 0; xx < AW; ++xx) acc += sTmp[n * AW + xx] * sFx[m * AW + xx];
            enc_in[(size_t)b * K_ENC + ch * 256 + tid] = (__bf16)(g * acc);
        }
    }
}

// ---------------- attention write: canvas += (Fy^T w Fx) / (gamma + 1e-4) ----------------
__global__ void write_kernel(const float* __restrict__ wbuf, const float* __restrict__ Fx,
                             const float* __restrict__ Fy, const float* __restrict__ gam,
                             float* __restrict__ canvas) {
    __shared__ float sFx[NPATCH * AW];
    __shared__ float sFy[NPATCH * AW];
    __shared__ float sW[NPATCH * NPATCH];
    __shared__ float sTmp[NPATCH * AW];
    int b = blockIdx.x, tid = threadIdx.x;
    for (int i = tid; i < NPATCH * AW; i += 256) {
        sFx[i] = Fx[(size_t)b * NPATCH * AW + i];
        sFy[i] = Fy[(size_t)b * NPATCH * AW + i];
    }
    float inv = 1.0f / (gam[b] + 1e-4f);
    for (int ch = 0; ch < 3; ++ch) {
        __syncthreads();
        sW[tid] = wbuf[(size_t)b * N_WR + ch * 256 + tid];
        __syncthreads();
        for (int i = tid; i < NPATCH * AW; i += 256) {   // tmp[n][xc] = sum_m w[n][m]*Fx[m][xc]
            int n = i >> 5, xc = i & 31;
            float acc = 0.0f;
#pragma unroll
            for (int m = 0; m < NPATCH; ++m) acc += sW[n * NPATCH + m] * sFx[m * AW + xc];
            sTmp[i] = acc;
        }
        __syncthreads();
        size_t base = (size_t)b * IMGE + (size_t)ch * CHW;
        for (int i = tid; i < CHW; i += 256) {           // canv[y][xc] = sum_n Fy[n][y]*tmp[n][xc]
            int y = i >> 5, xc = i & 31;
            float acc = 0.0f;
#pragma unroll
            for (int n = 0; n < NPATCH; ++n) acc += sFy[n * AW + y] * sTmp[n * AW + xc];
            canvas[base + i] += acc * inv;
        }
    }
}

// ---------------- elementwise recurrent cells ----------------
__global__ void gru_kernel(const float* __restrict__ gx, const float* __restrict__ gh,
                           float* __restrict__ h_f, __bf16* __restrict__ h_b) {
    int idx = blockIdx.x * blockDim.x + threadIdx.x;
    if (idx >= BSZ * HDIM) return;
    int b = idx / HDIM, j = idx % HDIM;
    size_t o = (size_t)b * N_G3;
    float r  = sigf(gx[o + j]        + gh[o + j]);
    float zt = sigf(gx[o + 100 + j]  + gh[o + 100 + j]);
    float n  = tanhf(gx[o + 200 + j] + r * gh[o + 200 + j]);
    float h  = (1.0f - zt) * n + zt * h_f[idx];
    h_f[idx] = h;
    h_b[(size_t)b * K_H + j] = (__bf16)h;
}

__global__ void z_kernel(const float* __restrict__ mu_t, const float* __restrict__ lv_t,
                         const float* __restrict__ eps,
                         float* __restrict__ mu_out, float* __restrict__ lv_out,
                         __bf16* __restrict__ z_b) {
    int idx = blockIdx.x * blockDim.x + threadIdx.x;
    if (idx >= BSZ * HDIM) return;
    int b = idx / HDIM, j = idx % HDIM;
    float mu = fmaxf(mu_t[(size_t)b * N_FC + j], 0.0f);
    float lv = fmaxf(lv_t[(size_t)b * N_FC + j], 0.0f);
    mu_out[idx] = mu;
    lv_out[idx] = lv;
    float z = eps[idx] * expf(0.5f * lv) + mu;
    z_b[(size_t)b * K_H + j] = (__bf16)z;
}

__global__ void lstm_kernel(const float* __restrict__ gx, const float* __restrict__ gh,
                            float* __restrict__ c_f, float* __restrict__ h_f,
                            __bf16* __restrict__ h_b) {
    int idx = blockIdx.x * blockDim.x + threadIdx.x;
    if (idx >= BSZ * HDIM) return;
    int b = idx / HDIM, j = idx % HDIM;
    size_t o = (size_t)b * N_G4;
    float gi = sigf(gx[o + j]         + gh[o + j]);
    float gf = sigf(gx[o + 100 + j]   + gh[o + 100 + j]);
    float gg = tanhf(gx[o + 200 + j]  + gh[o + 200 + j]);
    float go = sigf(gx[o + 300 + j]   + gh[o + 300 + j]);
    float c2 = gf * c_f[idx] + gi * gg;
    float h2 = go * tanhf(c2);
    c_f[idx] = c2;
    h_f[idx] = h2;
    h_b[(size_t)b * K_H + j] = (__bf16)h2;
}

__global__ void sigmoid_out_kernel(const float* __restrict__ canvas, float* __restrict__ out, int n) {
    int i = blockIdx.x * blockDim.x + threadIdx.x;
    if (i < n) out[i] = sigf(canvas[i]);
}

// ---------------- host launcher ----------------
extern "C" void kernel_launch(void* const* d_in, const int* in_sizes, int n_in,
                              void* d_out, int out_size, void* d_ws, size_t ws_size,
                              hipStream_t stream) {
    const float* x       = (const float*)d_in[0];
    const float* noise   = (const float*)d_in[1];
    const float* Wih_mu  = (const float*)d_in[2];
    const float* Whh_mu  = (const float*)d_in[3];
    const float* bih_mu  = (const float*)d_in[4];
    const float* bhh_mu  = (const float*)d_in[5];
    const float* Wih_lv  = (const float*)d_in[6];
    const float* Whh_lv  = (const float*)d_in[7];
    const float* bih_lv  = (const float*)d_in[8];
    const float* bhh_lv  = (const float*)d_in[9];
    const float* W_mu    = (const float*)d_in[10];
    const float* b_mu    = (const float*)d_in[11];
    const float* W_lv    = (const float*)d_in[12];
    const float* b_lv    = (const float*)d_in[13];
    const float* Wih_dec = (const float*)d_in[14];
    const float* Whh_dec = (const float*)d_in[15];
    const float* bih_dec = (const float*)d_in[16];
    const float* bhh_dec = (const float*)d_in[17];
    const float* W_att   = (const float*)d_in[18];
    const float* b_att   = (const float*)d_in[19];
    const float* W_wr    = (const float*)d_in[20];
    const float* b_wr    = (const float*)d_in[21];

    uint8_t* ws = (uint8_t*)d_ws;
    size_t off = 0;
    auto alloc = [&](size_t bytes) -> uint8_t* {
        uint8_t* p = ws + off;
        off = (off + bytes + 255) & ~(size_t)255;
        return p;
    };
    // bf16 weights (zero-padded)
    __bf16* Wb_ih_mu  = (__bf16*)alloc((size_t)N_G3 * K_ENC * 2);
    __bf16* Wb_ih_lv  = (__bf16*)alloc((size_t)N_G3 * K_ENC * 2);
    __bf16* Wb_hh_mu  = (__bf16*)alloc((size_t)N_G3 * K_H * 2);
    __bf16* Wb_hh_lv  = (__bf16*)alloc((size_t)N_G3 * K_H * 2);
    __bf16* Wb_mu     = (__bf16*)alloc((size_t)N_FC * K_H * 2);
    __bf16* Wb_lv     = (__bf16*)alloc((size_t)N_FC * K_H * 2);
    __bf16* Wb_ih_dec = (__bf16*)alloc((size_t)N_G4 * K_H * 2);
    __bf16* Wb_hh_dec = (__bf16*)alloc((size_t)N_G4 * K_H * 2);
    __bf16* Wb_wr     = (__bf16*)alloc((size_t)N_WR * K_H * 2);
    // state
    float*  canv      = (float*) alloc((size_t)BSZ * IMGE * 4);
    float*  h_mu_f    = (float*) alloc((size_t)BSZ * HDIM * 4);
    float*  h_lv_f    = (float*) alloc((size_t)BSZ * HDIM * 4);
    float*  h_dec_f   = (float*) alloc((size_t)BSZ * HDIM * 4);
    float*  c_dec_f   = (float*) alloc((size_t)BSZ * HDIM * 4);
    __bf16* h_mu_b    = (__bf16*)alloc((size_t)BSZ * K_H * 2);
    __bf16* h_lv_b    = (__bf16*)alloc((size_t)BSZ * K_H * 2);
    __bf16* h_dec_b   = (__bf16*)alloc((size_t)BSZ * K_H * 2);
    __bf16* z_b       = (__bf16*)alloc((size_t)BSZ * K_H * 2);
    __bf16* enc_in_b  = (__bf16*)alloc((size_t)BSZ * K_ENC * 2);
    float*  Fxb       = (float*) alloc((size_t)BSZ * NPATCH * AW * 4);
    float*  Fyb       = (float*) alloc((size_t)BSZ * NPATCH * AW * 4);
    float*  gamb      = (float*) alloc((size_t)BSZ * 4);
    float*  gx_mu     = (float*) alloc((size_t)BSZ * N_G3 * 4);
    float*  gh_mu     = (float*) alloc((size_t)BSZ * N_G3 * 4);
    float*  gx_lv     = (float*) alloc((size_t)BSZ * N_G3 * 4);
    float*  gh_lv     = (float*) alloc((size_t)BSZ * N_G3 * 4);
    float*  gdx       = (float*) alloc((size_t)BSZ * N_G4 * 4);
    float*  gdh       = (float*) alloc((size_t)BSZ * N_G4 * 4);
    float*  mu_tmp    = (float*) alloc((size_t)BSZ * N_FC * 4);
    float*  lv_tmp    = (float*) alloc((size_t)BSZ * N_FC * 4);
    float*  wbuf      = (float*) alloc((size_t)BSZ * N_WR * 4);
    (void)ws_size; (void)in_sizes; (void)n_in; (void)out_size;

    float* out        = (float*)d_out;
    float* out_canvas = out;                               // [2048,3,32,32]
    float* out_mu     = out + (size_t)BSZ * IMGE;          // [16,2048,100]
    float* out_lv     = out_mu + (size_t)T_STEPS * BSZ * HDIM;

    auto zero = [&](void* p, size_t u32_count) {
        int blocks = (int)((u32_count + 255) / 256);
        if (blocks > 4096) blocks = 4096;
        zero_u32_kernel<<<blocks, 256, 0, stream>>>((unsigned*)p, u32_count);
    };
    auto conv = [&](const float* s, __bf16* d, int N, int K, int Np, int Kp) {
        size_t tot = (size_t)Np * Kp;
        int blocks = (int)((tot + 255) / 256);
        if (blocks > 4096) blocks = 4096;
        convert_pad_kernel<<<blocks, 256, 0, stream>>>(s, d, N, K, Np, Kp);
    };
    auto gemm = [&](const __bf16* A, int lda, const __bf16* W, int ldw,
                    const float* bias, int Nreal, float* C, int Npad, int Kpad) {
        int tiles = (BSZ / 32) * (Npad / 32);   // 32x32 block per wave
        int blocks = (tiles + 7) / 8;
        wmma_gemm_kernel<<<blocks, 256, 0, stream>>>(A, lda, W, ldw, bias, Nreal,
                                                     C, Npad, BSZ, Npad, Kpad);
    };

    // weight conversion (once per launch)
    conv(Wih_mu,  Wb_ih_mu,  300, IN_ENC, N_G3, K_ENC);
    conv(Wih_lv,  Wb_ih_lv,  300, IN_ENC, N_G3, K_ENC);
    conv(Whh_mu,  Wb_hh_mu,  300, HDIM,   N_G3, K_H);
    conv(Whh_lv,  Wb_hh_lv,  300, HDIM,   N_G3, K_H);
    conv(W_mu,    Wb_mu,     100, HDIM,   N_FC, K_H);
    conv(W_lv,    Wb_lv,     100, HDIM,   N_FC, K_H);
    conv(Wih_dec, Wb_ih_dec, 400, HDIM,   N_G4, K_H);
    conv(Whh_dec, Wb_hh_dec, 400, HDIM,   N_G4, K_H);
    conv(W_wr,    Wb_wr,     768, HDIM,   N_WR, K_H);

    // zero state (every launch: deterministic, ws is poisoned)
    zero(canv,    (size_t)BSZ * IMGE);
    zero(h_mu_f,  (size_t)BSZ * HDIM);
    zero(h_lv_f,  (size_t)BSZ * HDIM);
    zero(h_dec_f, (size_t)BSZ * HDIM);
    zero(c_dec_f, (size_t)BSZ * HDIM);
    zero(h_mu_b,  (size_t)BSZ * K_H / 2);
    zero(h_lv_b,  (size_t)BSZ * K_H / 2);
    zero(h_dec_b, (size_t)BSZ * K_H / 2);
    zero(z_b,     (size_t)BSZ * K_H / 2);
    zero(enc_in_b,(size_t)BSZ * K_ENC / 2);

    const int EW_BLOCKS = (BSZ * HDIM + 255) / 256;
    for (int t = 0; t < T_STEPS; ++t) {
        // read-phase attention (old h_dec)
        attn_kernel<<<BSZ / 256, 256, 0, stream>>>(h_dec_f, W_att, b_att, Fxb, Fyb, gamb);
        read_kernel<<<BSZ, 256, 0, stream>>>(x, canv, Fxb, Fyb, gamb, h_dec_f, enc_in_b);
        // GRU gates (big WMMA GEMMs)
        gemm(enc_in_b, K_ENC, Wb_ih_mu, K_ENC, bih_mu, 300, gx_mu, N_G3, K_ENC);
        gemm(h_mu_b,   K_H,   Wb_hh_mu, K_H,   bhh_mu, 300, gh_mu, N_G3, K_H);
        gemm(enc_in_b, K_ENC, Wb_ih_lv, K_ENC, bih_lv, 300, gx_lv, N_G3, K_ENC);
        gemm(h_lv_b,   K_H,   Wb_hh_lv, K_H,   bhh_lv, 300, gh_lv, N_G3, K_H);
        gru_kernel<<<EW_BLOCKS, 256, 0, stream>>>(gx_mu, gh_mu, h_mu_f, h_mu_b);
        gru_kernel<<<EW_BLOCKS, 256, 0, stream>>>(gx_lv, gh_lv, h_lv_f, h_lv_b);
        // FC heads + reparameterization
        gemm(h_mu_b, K_H, Wb_mu, K_H, b_mu, 100, mu_tmp, N_FC, K_H);
        gemm(h_lv_b, K_H, Wb_lv, K_H, b_lv, 100, lv_tmp, N_FC, K_H);
        z_kernel<<<EW_BLOCKS, 256, 0, stream>>>(mu_tmp, lv_tmp,
                                                noise + (size_t)t * BSZ * HDIM,
                                                out_mu + (size_t)t * BSZ * HDIM,
                                                out_lv + (size_t)t * BSZ * HDIM, z_b);
        // decoder LSTM
        gemm(z_b,     K_H, Wb_ih_dec, K_H, bih_dec, 400, gdx, N_G4, K_H);
        gemm(h_dec_b, K_H, Wb_hh_dec, K_H, bhh_dec, 400, gdh, N_G4, K_H);
        lstm_kernel<<<EW_BLOCKS, 256, 0, stream>>>(gdx, gdh, c_dec_f, h_dec_f, h_dec_b);
        // write-phase attention (new h_dec) + write head + canvas update
        attn_kernel<<<BSZ / 256, 256, 0, stream>>>(h_dec_f, W_att, b_att, Fxb, Fyb, gamb);
        gemm(h_dec_b, K_H, Wb_wr, K_H, b_wr, 768, wbuf, N_WR, K_H);
        write_kernel<<<BSZ, 256, 0, stream>>>(wbuf, Fxb, Fyb, gamb, canv);
    }
    // final output canvas
    int n = BSZ * IMGE;
    sigmoid_out_kernel<<<(n + 255) / 256, 256, 0, stream>>>(canv, out_canvas, n);
}
